// Encoder_35854386987418
// MI455X (gfx1250) — compile-verified, matrix operand order checked by reading, not run on hardware
//
#include <hip/hip_runtime.h>
#include <hip/hip_bf16.h>
#include <stdint.h>

// CDNA5 (gfx1250) wave32 transformer encoder.
// - All GEMMs via V_WMMA_F32_16X16X32_BF16 (f32 accumulate).
// - bf16 operands streamed to LDS with GLOBAL_LOAD_ASYNC_TO_LDS_B128,
//   double-buffered and throttled with s_wait_asynccnt (in-order completion).
// - Weights converted (and transposed to N-major) once per layer.

typedef __attribute__((ext_vector_type(16))) __bf16 v16bf;
typedef __attribute__((ext_vector_type(8)))  float  v8f;

namespace {

constexpr int cB = 8, cS = 512, cD = 1024, cH = 16, cDK = 64, cDV = 64;
constexpr int cDFF = 4096, cL = 4;
constexpr long long X_ELEMS    = (long long)cB * cS * cD;          // 4,194,304
constexpr long long ATTN_ELEMS = (long long)cB * cH * cS * cS;     // 33,554,432

constexpr int BM = 128, BN = 128, BK = 32;
constexpr int LDSS = 40;   // u16 row stride: 80B rows -> b128-aligned, banks spread

enum { EP_NONE = 0, EP_BIAS = 1, EP_RELU = 2, EP_SCALEMASK = 4 };

__device__ __forceinline__ uint16_t bfbits(float x) {
  union { __bf16 h; uint16_t u; } c; c.h = (__bf16)x; return c.u;
}
__device__ __forceinline__ uint32_t pkbf(float a, float b) {
  union { __bf16 h[2]; uint32_t u; } p;
  p.h[0] = (__bf16)a; p.h[1] = (__bf16)b; return p.u;
}

// Async global->LDS 16-byte copy per lane (gfx1250, tracked by ASYNCcnt).
__device__ __forceinline__ void async_g2l_b128(const uint16_t* lds_dst, const uint16_t* gsrc) {
  uint32_t loff = (uint32_t)(uintptr_t)lds_dst;   // low 32 bits = LDS byte offset
  asm volatile("global_load_async_to_lds_b128 %0, %1, off"
               :: "v"(loff), "v"(gsrc) : "memory");
}
__device__ __forceinline__ void wait_async0() {
  asm volatile("s_wait_asynccnt 0" ::: "memory");
}
__device__ __forceinline__ void wait_async4() {   // oldest 4 copies retired
  asm volatile("s_wait_asynccnt 4" ::: "memory");
}

// ---------------------------------------------------------------------------
// WMMA GEMM:  C[z] = epilogue( A[z](MxK) * B[z](KxN) )
//   A: bf16 row-major (M x K)  [CONVA: f32 row-major, converted while staging]
//   B: bf16 N-major  (B[k][n] = Bs[n*ldb + k])  -- always
//   OUTM: 0 = f32 row-major, 1 = bf16 row-major, 2 = bf16 transposed V-store
//   batch offset: off = (z/hdiv)*s?1 + (z%hdiv)*s?2
// ---------------------------------------------------------------------------
template <int EPIL, bool CONVA, int OUTM>
__global__ __launch_bounds__(256) void gemm_wmma(
    const void* __restrict__ Araw, int lda, long long sA1, long long sA2,
    const uint16_t* __restrict__ Bs, int ldb, long long sB1, long long sB2,
    void* __restrict__ Craw, int ldc, long long sC1, long long sC2,
    int M, int N, int K, int hdiv,
    const float* __restrict__ bias, float scale,
    const int* __restrict__ tokens)
{
  constexpr int NBUF = CONVA ? 1 : 2;               // double buffer async path
  __shared__ uint16_t Als[NBUF][BM][LDSS];
  __shared__ uint16_t Bls[NBUF][BN][LDSS];

  const int z  = blockIdx.z;
  const int zq = z / hdiv;
  const int zr = z - zq * hdiv;
  const long long offA = zq * sA1 + zr * sA2;
  const uint16_t* Ab = (const uint16_t*)Araw + offA;
  const float*    Af = (const float*)Araw + offA;
  const uint16_t* Bp = Bs + zq * sB1 + zr * sB2;

  const int m0 = blockIdx.y * BM;
  const int n0 = blockIdx.x * BN;
  const int tid  = threadIdx.x;
  const int lane = tid & 31;
  const int wave = tid >> 5;
  const int wm = (wave & 1) * 64;
  const int wn = (wave >> 1) * 32;
  const int ln16  = lane & 15;
  const int khalf = lane >> 4;

  v8f zero = {0.f, 0.f, 0.f, 0.f, 0.f, 0.f, 0.f, 0.f};
  v8f acc[4][2];
#pragma unroll
  for (int i = 0; i < 4; ++i)
#pragma unroll
    for (int j = 0; j < 2; ++j) acc[i][j] = zero;

  // -- staging helpers ------------------------------------------------------
  auto stageA_async = [&](int k0, int pb) {
#pragma unroll
    for (int i = 0; i < 2; ++i) {
      int c   = tid + i * 256;                       // 512 b128 chunks = 128x32
      int row = c >> 2;
      int o8  = (c & 3) << 3;
      async_g2l_b128(&Als[pb][row][o8], Ab + (long long)(m0 + row) * lda + k0 + o8);
    }
  };
  auto stageB_async = [&](int k0, int pb) {
#pragma unroll
    for (int i = 0; i < 2; ++i) {
      int c   = tid + i * 256;
      int row = c >> 2;
      int o8  = (c & 3) << 3;
      async_g2l_b128(&Bls[pb][row][o8], Bp + (long long)(n0 + row) * ldb + k0 + o8);
    }
  };
  auto stageA_conv = [&](int k0) {                   // f32 -> bf16 while staging
#pragma unroll
    for (int i = 0; i < 4; ++i) {
      int lin = tid + i * 256;                       // 1024 float4 = 128x32 f32
      int row = lin >> 3;
      int kq  = (lin & 7) << 2;
      const float4 fv = *reinterpret_cast<const float4*>(
          Af + (long long)(m0 + row) * lda + k0 + kq);
      *reinterpret_cast<uint32_t*>(&Als[0][row][kq])     = pkbf(fv.x, fv.y);
      *reinterpret_cast<uint32_t*>(&Als[0][row][kq + 2]) = pkbf(fv.z, fv.w);
    }
  };
  auto compute = [&](int pb) {
    union Frag { uint32_t u[8]; v16bf v; };
    Frag af[4], bf[2];
#pragma unroll
    for (int ms = 0; ms < 4; ++ms) {
      int r = wm + ms * 16 + ln16;
#pragma unroll
      for (int v = 0; v < 8; ++v) {
        int kp = (v < 4) ? (khalf * 8 + 2 * v) : (16 + khalf * 8 + 2 * (v - 4));
        af[ms].u[v] = *reinterpret_cast<const uint32_t*>(&Als[pb][r][kp]);
      }
    }
#pragma unroll
    for (int ns = 0; ns < 2; ++ns) {
      int cidx = wn + ns * 16 + ln16;
#pragma unroll
      for (int v = 0; v < 8; ++v) {
        int kp = (v < 4) ? (khalf * 8 + 2 * v) : (16 + khalf * 8 + 2 * (v - 4));
        bf[ns].u[v] = *reinterpret_cast<const uint32_t*>(&Bls[pb][cidx][kp]);
      }
    }
#pragma unroll
    for (int ms = 0; ms < 4; ++ms)
#pragma unroll
      for (int ns = 0; ns < 2; ++ns)
        acc[ms][ns] = __builtin_amdgcn_wmma_f32_16x16x32_bf16(
            false, af[ms].v, false, bf[ns].v, (short)0, acc[ms][ns], false, false);
  };

  // -- main loop ------------------------------------------------------------
  if (CONVA) {
    for (int k0 = 0; k0 < K; k0 += BK) {
      stageA_conv(k0);
      stageB_async(k0, 0);
      if (k0 + BK < K)
        __builtin_prefetch(Af + (long long)(m0 + (tid >> 1)) * lda + k0 + BK + (tid & 1) * 16, 0, 0);
      wait_async0();
      __syncthreads();
      compute(0);
      __syncthreads();
    }
  } else {
    const int nk = K / BK;
    stageA_async(0, 0);
    stageB_async(0, 0);
    for (int kt = 0; kt < nk; ++kt) {
      const int pb = kt & 1;
      const bool more = (kt + 1 < nk);
      if (more) {                                    // stream tile kt+1 now
        stageA_async((kt + 1) * BK, pb ^ 1);
        stageB_async((kt + 1) * BK, pb ^ 1);
        wait_async4();                               // tile kt's 4 copies done
      } else {
        wait_async0();
      }
      __syncthreads();                               // tile kt visible to all
      compute(pb);
      __syncthreads();                               // done reading buf pb
    }
  }

  // -- epilogue -------------------------------------------------------------
  float*    Cf = (float*)Craw    + (zq * sC1 + zr * sC2);
  uint16_t* Ch = (uint16_t*)Craw + (zq * sC1 + zr * sC2);
#pragma unroll
  for (int ms = 0; ms < 4; ++ms) {
#pragma unroll
    for (int ns = 0; ns < 2; ++ns) {
      int col = n0 + wn + ns * 16 + ln16;
      if (col >= N) continue;
      float bval = 0.f;
      if (EPIL & EP_BIAS) bval = bias[col];
      bool padc = false;
      if (EPIL & EP_SCALEMASK) padc = (tokens[(long long)zq * cS + col] == 0);
#pragma unroll
      for (int r = 0; r < 8; ++r) {
        int row = m0 + wm + ms * 16 + khalf * 8 + r;
        if (row >= M) continue;
        float v = acc[ms][ns][r];
        if (EPIL & EP_SCALEMASK) v = padc ? -1e9f : v * scale;
        v += bval;
        if (EPIL & EP_RELU) v = fmaxf(v, 0.f);
        if (OUTM == 0) {
          Cf[(long long)row * ldc + col] = v;
        } else if (OUTM == 1) {
          Ch[(long long)row * ldc + col] = bfbits(v);
        } else {  // V-transposed: out[b][col][s], b=row>>9, s=row&511
          Ch[((long long)(row >> 9) << 19) + ((long long)col << 9) + (row & 511)] = bfbits(v);
        }
      }
    }
  }
}

// ---------------------------------------------------------------------------
__global__ __launch_bounds__(256) void softmax512(float* __restrict__ p) {
  __shared__ float red[256];
  float* row = p + (long long)blockIdx.x * 512;
  int t = threadIdx.x;
  float a0 = row[t], a1 = row[t + 256];
  red[t] = fmaxf(a0, a1);
  __syncthreads();
  for (int s = 128; s > 0; s >>= 1) { if (t < s) red[t] = fmaxf(red[t], red[t + s]); __syncthreads(); }
  float m = red[0];
  __syncthreads();
  float e0 = __expf(a0 - m), e1 = __expf(a1 - m);
  red[t] = e0 + e1;
  __syncthreads();
  for (int s = 128; s > 0; s >>= 1) { if (t < s) red[t] += red[t + s]; __syncthreads(); }
  float inv = 1.0f / red[0];
  row[t] = e0 * inv;
  row[t + 256] = e1 * inv;
}

// out = LayerNorm(a + b)*g + be ; optional bf16 mirror in obf
__global__ __launch_bounds__(256) void add_ln(
    const float* __restrict__ a, const float* __restrict__ b,
    const float* __restrict__ g, const float* __restrict__ be,
    float* __restrict__ o, uint16_t* __restrict__ obf)
{
  __shared__ float red[256];
  long long base = (long long)blockIdx.x * cD;
  int t = threadIdx.x;
  float x[4]; float s = 0.f;
#pragma unroll
  for (int i = 0; i < 4; ++i) { x[i] = a[base + t + i * 256] + b[base + t + i * 256]; s += x[i]; }
  red[t] = s; __syncthreads();
  for (int q = 128; q > 0; q >>= 1) { if (t < q) red[t] += red[t + q]; __syncthreads(); }
  float mean = red[0] * (1.0f / cD);
  __syncthreads();
  float vs = 0.f;
#pragma unroll
  for (int i = 0; i < 4; ++i) { float d = x[i] - mean; vs += d * d; }
  red[t] = vs; __syncthreads();
  for (int q = 128; q > 0; q >>= 1) { if (t < q) red[t] += red[t + q]; __syncthreads(); }
  float inv = rsqrtf(red[0] * (1.0f / cD) + 1e-6f);
#pragma unroll
  for (int i = 0; i < 4; ++i) {
    int c = t + i * 256;
    float r = (x[i] - mean) * inv * g[c] + be[c];
    o[base + c] = r;
    if (obf) obf[base + c] = bfbits(r);
  }
}

// x[b,s,:] = emb[token] + sinusoid(pos), plus bf16 mirror
__global__ __launch_bounds__(256) void embed_pos(
    const int* __restrict__ tok, const float* __restrict__ emb,
    float* __restrict__ x, uint16_t* __restrict__ xb)
{
  int bs = blockIdx.x;
  int s  = bs & (cS - 1);
  int token = tok[bs];
  int pos = (token == 0) ? 0 : (s + 1);
  long long base  = (long long)bs * cD;
  long long ebase = (long long)token * cD;
  for (int i = threadIdx.x; i < cD; i += 256) {
    double ang = (double)pos / pow(10000.0, (double)(2 * (i / 2)) / (double)cD);
    float pv = (i & 1) ? (float)cos(ang) : (float)sin(ang);
    float r = emb[ebase + i] + pv;
    x[base + i]  = r;
    xb[base + i] = bfbits(r);
  }
}

// straight f32 -> bf16 (layout preserved)
__global__ __launch_bounds__(256) void conv_bf16(
    const float* __restrict__ in, uint16_t* __restrict__ o, long long n)
{
  long long i = ((long long)blockIdx.x * 256 + threadIdx.x) * 4;
  if (i + 3 < n) {
    float4 f = *reinterpret_cast<const float4*>(in + i);
    uint2 st; st.x = pkbf(f.x, f.y); st.y = pkbf(f.z, f.w);
    *reinterpret_cast<uint2*>(o + i) = st;
  }
}

// transpose + convert: o[n*R + k] = bf16(in[k*Ncols + n])
__global__ __launch_bounds__(256) void convT_bf16(
    const float* __restrict__ in, uint16_t* __restrict__ o, int R, int Ncols)
{
  __shared__ uint16_t t[32][33];
  int nb = blockIdx.x * 32, kb = blockIdx.y * 32;
  int tx = threadIdx.x & 31, ty = threadIdx.x >> 5;   // ty 0..7
#pragma unroll
  for (int i = 0; i < 4; ++i) {
    int k = kb + ty * 4 + i;
    t[tx][ty * 4 + i] = bfbits(in[(long long)k * Ncols + nb + tx]);
  }
  __syncthreads();
#pragma unroll
  for (int i = 0; i < 4; ++i) {
    int nloc = ty * 4 + i;
    o[(long long)(nb + nloc) * R + kb + tx] = t[nloc][tx];
  }
}

} // namespace

extern "C" void kernel_launch(void* const* d_in, const int* in_sizes, int n_in,
                              void* d_out, int out_size, void* d_ws, size_t ws_size,
                              hipStream_t stream)
{
  const int*   tok   = (const int*)  d_in[0];
  const float* emb   = (const float*)d_in[1];
  const float* wq    = (const float*)d_in[2];
  const float* wk    = (const float*)d_in[3];
  const float* wv    = (const float*)d_in[4];
  const float* wo    = (const float*)d_in[5];
  const float* w1    = (const float*)d_in[6];
  const float* w2    = (const float*)d_in[7];
  const float* bq    = (const float*)d_in[8];
  const float* bk    = (const float*)d_in[9];
  const float* bv    = (const float*)d_in[10];
  const float* bo    = (const float*)d_in[11];
  const float* b1    = (const float*)d_in[12];
  const float* b2    = (const float*)d_in[13];
  const float* g_mha = (const float*)d_in[14];
  const float* b_mha = (const float*)d_in[15];
  const float* g_ffn = (const float*)d_in[16];
  const float* b_ffn = (const float*)d_in[17];
  const float* g_lay = (const float*)d_in[18];
  const float* b_lay = (const float*)d_in[19];
  float* out = (float*)d_out;
  float* ws  = (float*)d_ws;

  // f32 region: 64 MB
  float* X  = ws;                 // activations (residual source)
  float* T0 = ws + 1 * X_ELEMS;   // raw gemm outputs
  float* T1 = ws + 2 * X_ELEMS;   // LN intermediates
  float* Hf = ws + 3 * X_ELEMS;   // post-attn hidden (residual source)
  // bf16 region: 104 MB
  uint16_t* ub  = (uint16_t*)(ws + 4 * X_ELEMS);
  uint16_t* Xb  = ub;                     // x (bf16)
  uint16_t* Qb  = ub + 1 * X_ELEMS;       // q proj
  uint16_t* Kbb = ub + 2 * X_ELEMS;       // k proj
  uint16_t* Vbt = ub + 3 * X_ELEMS;       // v proj, transposed [b][h*64+d][s]
  uint16_t* Cb  = ub + 4 * X_ELEMS;       // ctx
  uint16_t* Hbb = ub + 5 * X_ELEMS;       // h (bf16)
  uint16_t* Wqb = ub + 6 * X_ELEMS;       // 1M each, N-major
  uint16_t* Wkb = Wqb + 1048576;
  uint16_t* Wvb = Wkb + 1048576;
  uint16_t* Wob = Wvb + 1048576;
  uint16_t* W1b = Wob + 1048576;          // 4M (already N-major)
  uint16_t* W2b = W1b + 4 * 1048576;      // 4M (already N-major)
  uint16_t* F1b = W2b + 4 * 1048576;      // 16M ffn hidden

  const int MS = cB * cS;                 // 4096
  const int HD = cH * cDK;                // 1024
  dim3 blk(256);

  embed_pos<<<dim3(MS), blk, 0, stream>>>(tok, emb, X, Xb);

  for (int l = 0; l < cL; ++l) {
    const float* wql = wq + (long long)l * cD * HD;
    const float* wkl = wk + (long long)l * cD * HD;
    const float* wvl = wv + (long long)l * cD * HD;
    const float* wol = wo + (long long)l * HD * cD;
    const float* w1l = w1 + (long long)l * cDFF * cD;
    const float* w2l = w2 + (long long)l * cD * cDFF;
    float* attn = out + X_ELEMS + (long long)l * ATTN_ELEMS;

    // weights -> bf16 (QKV/WO also transposed to N-major)
    convT_bf16<<<dim3(32, 32), blk, 0, stream>>>(wql, Wqb, cD, HD);
    convT_bf16<<<dim3(32, 32), blk, 0, stream>>>(wkl, Wkb, cD, HD);
    convT_bf16<<<dim3(32, 32), blk, 0, stream>>>(wvl, Wvb, cD, HD);
    convT_bf16<<<dim3(32, 32), blk, 0, stream>>>(wol, Wob, HD, cD);
    conv_bf16<<<dim3(4096), blk, 0, stream>>>(w1l, W1b, (long long)cDFF * cD);
    conv_bf16<<<dim3(4096), blk, 0, stream>>>(w2l, W2b, (long long)cD * cDFF);

    // Q/K/V projections (bf16 in, bf16 out; V stored transposed)
    gemm_wmma<EP_BIAS, false, 1><<<dim3(HD / BN, MS / BM, 1), blk, 0, stream>>>(
        Xb, cD, 0, 0, Wqb, cD, 0, 0, Qb, HD, 0, 0,
        MS, HD, cD, 1, bq + (long long)l * HD, 1.f, nullptr);
    gemm_wmma<EP_BIAS, false, 1><<<dim3(HD / BN, MS / BM, 1), blk, 0, stream>>>(
        Xb, cD, 0, 0, Wkb, cD, 0, 0, Kbb, HD, 0, 0,
        MS, HD, cD, 1, bk + (long long)l * HD, 1.f, nullptr);
    gemm_wmma<EP_BIAS, false, 2><<<dim3(HD / BN, MS / BM, 1), blk, 0, stream>>>(
        Xb, cD, 0, 0, Wvb, cD, 0, 0, Vbt, cS, 0, 0,
        MS, HD, cD, 1, bv + (long long)l * HD, 1.f, nullptr);

    // scores = Q K^T / 8, pad-mask -> f32 straight into d_out attn slot
    gemm_wmma<EP_SCALEMASK, false, 0><<<dim3(cS / BN, cS / BM, cB * cH), blk, 0, stream>>>(
        Qb,  HD, (long long)cS * HD, (long long)cDK,
        Kbb, HD, (long long)cS * HD, (long long)cDK,
        attn, cS, (long long)cH * cS * cS, (long long)cS * cS,
        cS, cS, cDK, cH, nullptr, 0.125f, tok);

    softmax512<<<dim3(cB * cH * cS), blk, 0, stream>>>(attn);

    // ctx = attn * V   (A converted f32->bf16 while staging; B = Vbt N-major)
    gemm_wmma<EP_NONE, true, 1><<<dim3(1, cS / BM, cB * cH), blk, 0, stream>>>(
        attn, cS, (long long)cH * cS * cS, (long long)cS * cS,
        Vbt, cS, (long long)1 << 19, (long long)cDV * cS,
        Cb,  HD, (long long)cS * HD, (long long)cDK,
        cS, cDV, cS, cH, nullptr, 1.f, nullptr);

    // mha_raw = ctx * Wo + bo -> T0 (f32)
    gemm_wmma<EP_BIAS, false, 0><<<dim3(cD / BN, MS / BM, 1), blk, 0, stream>>>(
        Cb, HD, 0, 0, Wob, HD, 0, 0, T0, cD, 0, 0,
        MS, cD, HD, 1, bo + (long long)l * cD, 1.f, nullptr);

    add_ln<<<dim3(MS), blk, 0, stream>>>(T0, X, g_mha + (long long)l * cD,
                                         b_mha + (long long)l * cD, T1, nullptr);
    add_ln<<<dim3(MS), blk, 0, stream>>>(T1, X, g_lay + (long long)l * cD,
                                         b_lay + (long long)l * cD, Hf, Hbb);

    // f1 = relu(h * W1^T + b1) -> bf16
    gemm_wmma<EP_BIAS | EP_RELU, false, 1><<<dim3(cDFF / BN, MS / BM, 1), blk, 0, stream>>>(
        Hbb, cD, 0, 0, W1b, cD, 0, 0, F1b, cDFF, 0, 0,
        MS, cDFF, cD, 1, b1 + (long long)l * cDFF, 1.f, nullptr);

    // f_raw = f1 * W2^T + b2 -> T0 (f32)
    gemm_wmma<EP_BIAS, false, 0><<<dim3(cD / BN, MS / BM, 1), blk, 0, stream>>>(
        F1b, cDFF, 0, 0, W2b, cDFF, 0, 0, T0, cD, 0, 0,
        MS, cD, cDFF, 1, b2 + (long long)l * cD, 1.f, nullptr);

    add_ln<<<dim3(MS), blk, 0, stream>>>(T0, Hf, g_ffn + (long long)l * cD,
                                         b_ffn + (long long)l * cD, T1, nullptr);
    add_ln<<<dim3(MS), blk, 0, stream>>>(T1, Hf, g_lay + (long long)l * cD,
                                         b_lay + (long long)l * cD,
                                         (l == cL - 1) ? out : X,
                                         (l == cL - 1) ? nullptr : Xb);
  }
  (void)in_sizes; (void)n_in; (void)out_size; (void)ws_size;
}
